// KPConvFPN_decoder_39633958207498
// MI455X (gfx1250) — compile-verified, matrix operand order checked by reading, not run on hardware
//
#include <hip/hip_runtime.h>
#include <stdint.h>
#include <stddef.h>

// ---------- vector types (plain ext-vectors: POD, union-safe) ----------
typedef __attribute__((ext_vector_type(16))) __bf16   v16bf;
typedef __attribute__((ext_vector_type(8)))  float    v8f;
typedef __attribute__((ext_vector_type(4)))  unsigned u32x4;
typedef __attribute__((ext_vector_type(2)))  unsigned u32x2;
typedef __attribute__((ext_vector_type(4)))  float    f32x4;

union FragBF { v16bf v; u32x4 u[2]; };

__device__ __forceinline__ unsigned short f32_to_bf16_bits(float f) {
  union { float f; unsigned u; } c; c.f = f;
  unsigned u = c.u;
  unsigned r = u + 0x7FFFu + ((u >> 16) & 1u);   // round-to-nearest-even
  return (unsigned short)(r >> 16);
}

// ---------- f32 -> bf16 bulk convert (4 elems / thread) ----------
__global__ __launch_bounds__(256)
void cvt_f32_to_bf16(const float* __restrict__ in, unsigned* __restrict__ out, size_t n4) {
  size_t i = (size_t)blockIdx.x * blockDim.x + threadIdx.x;
  if (i >= n4) return;
  f32x4 v = *(const f32x4*)(in + i * 4);
  u32x2 p;
  p[0] = (unsigned)f32_to_bf16_bits(v[0]) | ((unsigned)f32_to_bf16_bits(v[1]) << 16);
  p[1] = (unsigned)f32_to_bf16_bits(v[2]) | ((unsigned)f32_to_bf16_bits(v[3]) << 16);
  *(u32x2*)(out + i * 2) = p;
}

// ---------- gather + concat -> bf16 A matrix ----------
// dst row i = [ prev[idx[i]] (C1 bf16 cols) | bf16(feats[i]) (C2 cols) ]
__global__ __launch_bounds__(256)
void build_concat_bf16(__bf16* __restrict__ dst, const __bf16* __restrict__ prev,
                       const int* __restrict__ idx, const float* __restrict__ feats,
                       int M, int C1, int C2) {
  const int Ctot = C1 + C2;
  const int chunks = Ctot >> 3;           // 8 cols (16B) per thread
  size_t t = (size_t)blockIdx.x * blockDim.x + threadIdx.x;
  if (t >= (size_t)M * (size_t)chunks) return;
  int row = (int)(t / chunks);
  int c8  = (int)(t % chunks) << 3;
  __bf16* drow = dst + (size_t)row * Ctot + c8;
  if (c8 < C1) {
    *(u32x4*)drow = *(const u32x4*)(prev + (size_t)idx[row] * C1 + c8);
  } else {
    const float* f = feats + (size_t)row * C2 + (c8 - C1);
    f32x4 f0 = *(const f32x4*)f;
    f32x4 f1 = *(const f32x4*)(f + 4);
    u32x4 p;
    p[0] = (unsigned)f32_to_bf16_bits(f0[0]) | ((unsigned)f32_to_bf16_bits(f0[1]) << 16);
    p[1] = (unsigned)f32_to_bf16_bits(f0[2]) | ((unsigned)f32_to_bf16_bits(f0[3]) << 16);
    p[2] = (unsigned)f32_to_bf16_bits(f1[0]) | ((unsigned)f32_to_bf16_bits(f1[1]) << 16);
    p[3] = (unsigned)f32_to_bf16_bits(f1[2]) | ((unsigned)f32_to_bf16_bits(f1[3]) << 16);
    *(u32x4*)drow = p;
  }
}

// ---------- WMMA GEMM: out[M,N] = A[M,K](bf16) * W[N,K](bf16)^T + bias ----------
// block = 128 threads (4 waves, 2x2), block tile 128x128, wave tile 64x64.
// Per 32-K chunk: 8 fragment loads (16 x b128) feed 16 WMMAs -> 1.0 load/wmma.
template <bool OUT_BF16>
__global__ __launch_bounds__(128)
void gemm_bf16_wmma(const __bf16* __restrict__ A, const __bf16* __restrict__ W,
                    const float* __restrict__ bias, void* __restrict__ out,
                    int M, int N, int K) {
  const int lane = threadIdx.x & 31;
  const int wave = threadIdx.x >> 5;
  const int lm = lane & 15;        // M (A-frag) / N (B-frag) row within tile
  const int kg = lane >> 4;        // K half select (ISA 16-bit layout)
  const int m0 = blockIdx.x * 128 + (wave >> 1) * 64;
  const int n0 = blockIdx.y * 128 + (wave & 1) * 64;

  v8f acc[4][4];
  #pragma unroll
  for (int i = 0; i < 4; ++i)
    #pragma unroll
    for (int j = 0; j < 4; ++j)
      #pragma unroll
      for (int r = 0; r < 8; ++r) acc[i][j][r] = 0.0f;

  const __bf16* ap[4];
  const __bf16* bp[4];
  #pragma unroll
  for (int i = 0; i < 4; ++i) ap[i] = A + (size_t)(m0 + i * 16 + lm) * K;
  #pragma unroll
  for (int j = 0; j < 4; ++j) bp[j] = W + (size_t)(n0 + j * 16 + lm) * K;

  for (int kk = 0; kk < K; kk += 32) {
    const int ko = kk + kg * 8;
    if (kk + 32 < K) {             // stream next A chunk -> global_prefetch
      __builtin_prefetch(ap[0] + ko + 32, 0, 1);
      __builtin_prefetch(ap[2] + ko + 32, 0, 1);
    }
    FragBF a[4], b[4];
    #pragma unroll
    for (int i = 0; i < 4; ++i) {
      a[i].u[0] = *(const u32x4*)(ap[i] + ko);
      a[i].u[1] = *(const u32x4*)(ap[i] + ko + 16);
    }
    #pragma unroll
    for (int j = 0; j < 4; ++j) {
      b[j].u[0] = *(const u32x4*)(bp[j] + ko);
      b[j].u[1] = *(const u32x4*)(bp[j] + ko + 16);
    }
    #pragma unroll
    for (int i = 0; i < 4; ++i)
      #pragma unroll
      for (int j = 0; j < 4; ++j)
        acc[i][j] = __builtin_amdgcn_wmma_f32_16x16x32_bf16(
            false, a[i].v, false, b[j].v, (short)0, acc[i][j], false, false);
  }

  // epilogue: C/D layout -> lane n = lm, VGPR r holds M = r + kg*8
  #pragma unroll
  for (int j = 0; j < 4; ++j) {
    const int n = n0 + j * 16 + lm;
    const float bv = bias[n];
    #pragma unroll
    for (int i = 0; i < 4; ++i) {
      const int mb = m0 + i * 16 + kg * 8;
      #pragma unroll
      for (int r = 0; r < 8; ++r) {
        float v = acc[i][j][r] + bv;
        size_t o = (size_t)(mb + r) * N + n;
        if (OUT_BF16) {
          union { unsigned short s; __bf16 b; } cb; cb.s = f32_to_bf16_bits(v);
          ((__bf16*)out)[o] = cb.b;
        } else {
          ((float*)out)[o] = v;
        }
      }
    }
  }
}

// ---------- GroupNorm (stats over C/g * N per group, matching reference) ----------
#define GN_SPLIT 64
__global__ __launch_bounds__(256)
void gn_reduce(const float* __restrict__ y, int M, int gsShift, float* __restrict__ partial) {
  const int g = blockIdx.x;                 // 0..31
  const int split = blockIdx.y;             // 0..GN_SPLIT-1
  const int gs = 1 << gsShift;
  const int C = 32 << gsShift;
  const size_t total = (size_t)M << gsShift;
  const size_t per = total / GN_SPLIT;      // M*gs divisible by 64 for both layers
  const size_t begin = (size_t)split * per;
  float s = 0.f, sq = 0.f;
  for (size_t e = begin + threadIdx.x; e < begin + per; e += 256) {
    size_t row = e >> gsShift;
    int col = (g << gsShift) + (int)(e & (size_t)(gs - 1));
    float v = y[row * (size_t)C + col];
    s += v; sq += v * v;
  }
  __shared__ float ls[256], lq[256];
  ls[threadIdx.x] = s; lq[threadIdx.x] = sq;
  __syncthreads();
  for (int off = 128; off > 0; off >>= 1) {
    if ((int)threadIdx.x < off) {
      ls[threadIdx.x] += ls[threadIdx.x + off];
      lq[threadIdx.x] += lq[threadIdx.x + off];
    }
    __syncthreads();
  }
  if (threadIdx.x == 0) {
    partial[(g * GN_SPLIT + split) * 2 + 0] = ls[0];
    partial[(g * GN_SPLIT + split) * 2 + 1] = lq[0];
  }
}

__global__ void gn_finalize(const float* __restrict__ partial, float* __restrict__ meanrstd,
                            float invCount) {
  int g = threadIdx.x;
  if (g >= 32) return;
  float s = 0.f, sq = 0.f;
  for (int i = 0; i < GN_SPLIT; ++i) {
    s  += partial[(g * GN_SPLIT + i) * 2 + 0];
    sq += partial[(g * GN_SPLIT + i) * 2 + 1];
  }
  float mean = s * invCount;
  float var  = sq * invCount - mean * mean;
  meanrstd[2 * g + 0] = mean;
  meanrstd[2 * g + 1] = rsqrtf(var + 1e-5f);
}

// normalize + affine + LeakyReLU(0.1) -> bf16
__global__ __launch_bounds__(256)
void gn_norm_act(const float* __restrict__ y, const float* __restrict__ meanrstd,
                 const float* __restrict__ gamma, const float* __restrict__ beta,
                 __bf16* __restrict__ out, size_t M, int C, int gsShift) {
  size_t total4 = (M * (size_t)C) >> 2;
  size_t i = (size_t)blockIdx.x * blockDim.x + threadIdx.x;
  if (i >= total4) return;
  size_t e = i << 2;
  int col = (int)(e % (size_t)C);           // 4-aligned; gs>=16 so same group
  int g = col >> gsShift;
  float mean = meanrstd[2 * g + 0];
  float rstd = meanrstd[2 * g + 1];
  f32x4 v = *(const f32x4*)(y + e);
  unsigned short h[4];
  #pragma unroll
  for (int r = 0; r < 4; ++r) {
    float x = (v[r] - mean) * rstd * gamma[col + r] + beta[col + r];
    x = (x >= 0.f) ? x : 0.1f * x;
    h[r] = f32_to_bf16_bits(x);
  }
  u32x2 p;
  p[0] = (unsigned)h[0] | ((unsigned)h[1] << 16);
  p[1] = (unsigned)h[2] | ((unsigned)h[3] << 16);
  *(u32x2*)(out + e) = p;
}

// ---------------------------- host side ----------------------------
extern "C" void kernel_launch(void* const* d_in, const int* in_sizes, int n_in,
                              void* d_out, int out_size, void* d_ws, size_t ws_size,
                              hipStream_t stream) {
  const float* feats_s2 = (const float*)d_in[0];
  const float* feats_s3 = (const float*)d_in[1];
  const float* feats_s4 = (const float*)d_in[2];
  const float* feats_s5 = (const float*)d_in[3];
  const int*   up1      = (const int*)  d_in[4];
  const int*   up2      = (const int*)  d_in[5];
  const int*   up3      = (const int*)  d_in[6];
  const float* W_in     = (const float*)d_in[7];
  const float* b_in     = (const float*)d_in[8];
  const float* W4       = (const float*)d_in[9];
  const float* b4       = (const float*)d_in[10];
  const float* g4       = (const float*)d_in[11];
  const float* bt4      = (const float*)d_in[12];
  const float* W3       = (const float*)d_in[13];
  const float* b3       = (const float*)d_in[14];
  const float* g3       = (const float*)d_in[15];
  const float* bt3      = (const float*)d_in[16];
  const float* W2       = (const float*)d_in[17];
  const float* b2       = (const float*)d_in[18];

  const int N5 = 2048, N4 = 8192, N3 = 32768, N2 = 131072;
  const int D5 = 2048;                 // feats_s5 width == inproj in-dim
  const int CIN = 2048;                // inproj out-dim
  const int C4i = 3072, C4o = 1024;    // decoder4
  const int C3i = 1536, C3o = 512;     // decoder3
  const int C2i = 768,  C2o = 256;     // decoder2

  // ---- workspace layout (lifetime-overlapped reuse; peak ~252 MB) ----
  char* ws = (char*)d_ws;
  const size_t OFF_WIN   = 0;                                   // 2048*2048 bf16
  const size_t OFF_W4    = OFF_WIN + (size_t)CIN * D5 * 2;      // 1024*3072 bf16
  const size_t OFF_W3    = OFF_W4  + (size_t)C4o * C4i * 2;     // 512*1536 bf16
  const size_t OFF_W2    = OFF_W3  + (size_t)C3o * C3i * 2;     // 256*768 bf16
  const size_t OFF_PART  = OFF_W2  + (size_t)C2o * C2i * 2;     // 32*64*2 f32
  const size_t OFF_STATS = OFF_PART + 32 * GN_SPLIT * 2 * 4;    // 32*2 f32
  const size_t BASE      = 16ull << 20;                         // activation region
  __bf16* Wb_in = (__bf16*)(ws + OFF_WIN);
  __bf16* Wb4   = (__bf16*)(ws + OFF_W4);
  __bf16* Wb3   = (__bf16*)(ws + OFF_W3);
  __bf16* Wb2   = (__bf16*)(ws + OFF_W2);
  float*  part  = (float*) (ws + OFF_PART);
  float*  stats = (float*) (ws + OFF_STATS);

  __bf16* x5bf = (__bf16*)(ws + BASE + 0);                      // 8 MB
  __bf16* A5bf = (__bf16*)(ws + BASE + 8388608ull);             // 8 MB (dead after GEMM1)
  __bf16* A4   = (__bf16*)(ws + BASE + 8388608ull);             // 48 MB (reuses A5)
  float*  y4   = (float*) (ws + BASE + 58720256ull);            // 32 MB
  __bf16* l4   = (__bf16*)(ws + BASE + 100663296ull);           // 16 MB
  __bf16* A3   = (__bf16*)(ws + BASE + 0);                      // 96 MB (reuses x5/A4/y4)
  float*  y3   = (float*) (ws + BASE + 117440512ull);           // 64 MB
  __bf16* l3   = (__bf16*)(ws + BASE + 0);                      // 32 MB (reuses A3)
  __bf16* A2   = (__bf16*)(ws + BASE + 33554432ull);            // 192 MB

  // 1) convert weights + feats_s5 to bf16
  auto cvt = [&](const float* src, void* dst, size_t n) {
    size_t n4 = n >> 2;
    cvt_f32_to_bf16<<<dim3((unsigned)((n4 + 255) / 256)), dim3(256), 0, stream>>>(
        src, (unsigned*)dst, n4);
  };
  cvt(W_in,     Wb_in, (size_t)CIN * D5);
  cvt(W4,       Wb4,   (size_t)C4o * C4i);
  cvt(W3,       Wb3,   (size_t)C3o * C3i);
  cvt(W2,       Wb2,   (size_t)C2o * C2i);
  cvt(feats_s5, A5bf,  (size_t)N5 * D5);

  // 2) x5 = feats_s5 @ W_in^T + b_in  -> bf16
  gemm_bf16_wmma<true><<<dim3(N5 / 128, CIN / 128), dim3(128), 0, stream>>>(
      A5bf, Wb_in, b_in, (void*)x5bf, N5, CIN, D5);

  // 3) A4 = [x5[up3] | bf16(feats_s4)]
  {
    size_t tot = (size_t)N4 * (C4i / 8);
    build_concat_bf16<<<dim3((unsigned)((tot + 255) / 256)), dim3(256), 0, stream>>>(
        A4, x5bf, up3, feats_s4, N4, CIN, 1024);
  }
  // 4) y4 = A4 @ W4^T + b4 (f32)
  gemm_bf16_wmma<false><<<dim3(N4 / 128, C4o / 128), dim3(128), 0, stream>>>(
      A4, Wb4, b4, (void*)y4, N4, C4o, C4i);
  // 5-7) GroupNorm + LeakyReLU -> l4 (bf16)
  gn_reduce<<<dim3(32, GN_SPLIT), dim3(256), 0, stream>>>(y4, N4, 5, part);
  gn_finalize<<<dim3(1), dim3(32), 0, stream>>>(part, stats, 1.0f / ((float)N4 * 32.0f));
  {
    size_t t4 = ((size_t)N4 * C4o) >> 2;
    gn_norm_act<<<dim3((unsigned)((t4 + 255) / 256)), dim3(256), 0, stream>>>(
        y4, stats, g4, bt4, l4, (size_t)N4, C4o, 5);
  }

  // 8) A3 = [l4[up2] | bf16(feats_s3)]
  {
    size_t tot = (size_t)N3 * (C3i / 8);
    build_concat_bf16<<<dim3((unsigned)((tot + 255) / 256)), dim3(256), 0, stream>>>(
        A3, l4, up2, feats_s3, N3, C4o, 512);
  }
  // 9) y3 = A3 @ W3^T + b3 (f32)
  gemm_bf16_wmma<false><<<dim3(N3 / 128, C3o / 128), dim3(128), 0, stream>>>(
      A3, Wb3, b3, (void*)y3, N3, C3o, C3i);
  // 10-12) GroupNorm + LeakyReLU -> l3 (bf16)
  gn_reduce<<<dim3(32, GN_SPLIT), dim3(256), 0, stream>>>(y3, N3, 4, part);
  gn_finalize<<<dim3(1), dim3(32), 0, stream>>>(part, stats, 1.0f / ((float)N3 * 16.0f));
  {
    size_t t4 = ((size_t)N3 * C3o) >> 2;
    gn_norm_act<<<dim3((unsigned)((t4 + 255) / 256)), dim3(256), 0, stream>>>(
        y3, stats, g3, bt3, l3, (size_t)N3, C3o, 4);
  }

  // 13) A2 = [l3[up1] | bf16(feats_s2)]
  {
    size_t tot = (size_t)N2 * (C2i / 8);
    build_concat_bf16<<<dim3((unsigned)((tot + 255) / 256)), dim3(256), 0, stream>>>(
        A2, l3, up1, feats_s2, N2, C3o, 256);
  }
  // 14) out = A2 @ W2^T + b2 (f32 -> d_out)
  gemm_bf16_wmma<false><<<dim3(N2 / 128, C2o / 128), dim3(128), 0, stream>>>(
      A2, Wb2, b2, d_out, N2, C2o, C2i);
}
